// OuterProduct_54692113547877
// MI455X (gfx1250) — compile-verified, matrix operand order checked by reading, not run on hardware
//
#include <hip/hip_runtime.h>

typedef __attribute__((ext_vector_type(16))) _Float16 v16h;
typedef __attribute__((ext_vector_type(8)))  _Float16 v8h;
typedef __attribute__((ext_vector_type(8)))  float    v8f;
typedef __attribute__((ext_vector_type(4)))  float    v4f;

#define H 128           // H1 == H2 == 128
#define L 1024
#define KNB 48          // neighbors per node
#define LN_EPS 1e-5f

static __device__ __forceinline__ v16h pack16(v8h lo, v8h hi) {
    v16h r;
#pragma unroll
    for (int j = 0; j < 8; ++j) { r[j] = lo[j]; r[j + 8] = hi[j]; }
    return r;
}

static __device__ __forceinline__ v8f wmma_acc(v16h a, v16h b, v8f c) {
    return __builtin_amdgcn_wmma_f32_16x16x32_f16(false, a, false, b,
                                                  (short)0, c, false, false);
}

// Split f32 matrices W1, W2 into f16 (hi, lo) pairs for split-precision WMMA.
__global__ __launch_bounds__(256) void split_w_kernel(
    const float* __restrict__ W1, const float* __restrict__ W2,
    _Float16* __restrict__ W1h, _Float16* __restrict__ W1l,
    _Float16* __restrict__ W2h, _Float16* __restrict__ W2l) {
    int t = blockIdx.x * 256 + threadIdx.x;   // 64 blocks * 256 = 16384 = H*H
    float a = W1[t];
    _Float16 ah = (_Float16)a;
    W1h[t] = ah; W1l[t] = (_Float16)(a - (float)ah);
    float b = W2[t];
    _Float16 bh = (_Float16)b;
    W2h[t] = bh; W2l[t] = (_Float16)(b - (float)bh);
}

// Kernel 1: hv = LayerNorm(h_V) @ W1^T + b1   (M=1024, N=128, K=128)
// One block = 16 rows. 8 waves: LN (2 rows/wave), then one 16x16 N-tile/wave.
__global__ __launch_bounds__(256) void ln_gemm1_kernel(
    const float* __restrict__ hV,
    const float* __restrict__ lnS, const float* __restrict__ lnB,
    const _Float16* __restrict__ W1h, const _Float16* __restrict__ W1l,
    const float* __restrict__ b1, float* __restrict__ hv) {
    __shared__ _Float16 lnh[16 * H];
    __shared__ _Float16 lnl[16 * H];

    const int tid  = threadIdx.x;
    const int wave = tid >> 5;
    const int lane = tid & 31;
    const int ibase = blockIdx.x * 16;

    // ---- LayerNorm: wave w handles local rows 2w, 2w+1; lane covers 4 h's.
#pragma unroll
    for (int rr = 0; rr < 2; ++rr) {
        const int lr = 2 * wave + rr;
        const int gi = ibase + lr;
        const int h0 = lane * 4;
        v4f x = *(const v4f*)(hV + (size_t)gi * H + h0);
        float s = x.x + x.y + x.z + x.w;
#pragma unroll
        for (int off = 16; off > 0; off >>= 1) s += __shfl_xor(s, off, 32);
        const float mu = s * (1.0f / H);
        v4f d; d.x = x.x - mu; d.y = x.y - mu; d.z = x.z - mu; d.w = x.w - mu;
        float q = d.x * d.x + d.y * d.y + d.z * d.z + d.w * d.w;
#pragma unroll
        for (int off = 16; off > 0; off >>= 1) q += __shfl_xor(q, off, 32);
        const float rs = __frsqrt_rn(q * (1.0f / H) + LN_EPS);
        v4f sc = *(const v4f*)(lnS + h0);
        v4f bi = *(const v4f*)(lnB + h0);
        float v[4] = { d.x * rs * sc.x + bi.x, d.y * rs * sc.y + bi.y,
                       d.z * rs * sc.z + bi.z, d.w * rs * sc.w + bi.w };
#pragma unroll
        for (int j = 0; j < 4; ++j) {
            _Float16 hi = (_Float16)v[j];
            lnh[lr * H + h0 + j] = hi;
            lnl[lr * H + h0 + j] = (_Float16)(v[j] - (float)hi);
        }
    }
    __syncthreads();

    // ---- GEMM tile: wave w -> output columns [16w, 16w+16)
    const int n     = lane & 15;
    const int o     = wave * 16 + n;
    const int khalf = (lane < 16) ? 0 : 8;

    v8f c = {};
#pragma unroll
    for (int c4 = 0; c4 < 4; ++c4) {
        const int c0 = 32 * c4 + khalf;
        // A (LN rows) from LDS, split layout per 16-bit A-matrix spec
        const int m = lane & 15;
        v16h Ah = pack16(*(const v8h*)(lnh + m * H + c0),
                         *(const v8h*)(lnh + m * H + c0 + 16));
        v16h Al = pack16(*(const v8h*)(lnl + m * H + c0),
                         *(const v8h*)(lnl + m * H + c0 + 16));
        // B = W1^T fragments: lane n reads two 8-element runs of W1 row o (c0.., c0+16..)
        v16h Bh = pack16(*(const v8h*)(W1h + (size_t)o * H + c0),
                         *(const v8h*)(W1h + (size_t)o * H + c0 + 16));
        v16h Bl = pack16(*(const v8h*)(W1l + (size_t)o * H + c0),
                         *(const v8h*)(W1l + (size_t)o * H + c0 + 16));
        c = wmma_acc(Ah, Bh, c);
        c = wmma_acc(Ah, Bl, c);
        c = wmma_acc(Al, Bh, c);
    }

    const float bias = b1[o];
    const int mofs = (lane < 16) ? 0 : 8;
#pragma unroll
    for (int r = 0; r < 8; ++r) {
        const int gi = ibase + r + mofs;
        hv[(size_t)gi * H + o] = c[r] + bias;
    }
}

// Kernel 2: out[i,k,o] = sum_h (hv[i,h]*hv[E[i,k],h]) * W2[o,h] + b2[o]
// One block per i. Panel A (48x128 outer products) split to f16 in LDS,
// each wave owns one 16-wide N-tile (B in registers) and 3 M-tiles.
__global__ __launch_bounds__(256) void gemm2_kernel(
    const float* __restrict__ hv, const int* __restrict__ Eidx,
    const _Float16* __restrict__ W2h, const _Float16* __restrict__ W2l,
    const float* __restrict__ b2, float* __restrict__ out) {
    __shared__ _Float16 Ah_s[KNB * H];   // 12 KB
    __shared__ _Float16 Al_s[KNB * H];   // 12 KB
    __shared__ float    hvi_s[H];
    __shared__ int      js[KNB];

    const int tid = threadIdx.x;
    const int i   = blockIdx.x;

    if (tid < H)   hvi_s[tid] = hv[(size_t)i * H + tid];
    if (tid < KNB) js[tid]    = Eidx[i * KNB + tid];
    __syncthreads();

    // Build split outer-product panel: A[k,h] = hv_i[h] * hv_j[h]
    for (int idx = tid; idx < KNB * H; idx += 256) {
        const int k = idx >> 7;           // neighbor slot
        const int h = idx & (H - 1);
        const float p = hvi_s[h] * hv[(size_t)js[k] * H + h];
        const _Float16 hi = (_Float16)p;
        Ah_s[idx] = hi;
        Al_s[idx] = (_Float16)(p - (float)hi);
    }
    __syncthreads();

    const int wave  = tid >> 5;
    const int lane  = tid & 31;
    const int n     = lane & 15;
    const int o     = wave * 16 + n;
    const int khalf = (lane < 16) ? 0 : 8;

    // Preload B (W2 split) fragments for all 4 K-chunks: stays in registers.
    v16h Bh[4], Bl[4];
#pragma unroll
    for (int c4 = 0; c4 < 4; ++c4) {
        const int c0 = 32 * c4 + khalf;
        Bh[c4] = pack16(*(const v8h*)(W2h + (size_t)o * H + c0),
                        *(const v8h*)(W2h + (size_t)o * H + c0 + 16));
        Bl[c4] = pack16(*(const v8h*)(W2l + (size_t)o * H + c0),
                        *(const v8h*)(W2l + (size_t)o * H + c0 + 16));
    }

    const float bias = b2[o];
    const int   m    = lane & 15;
    const int   mofs = (lane < 16) ? 0 : 8;
    const size_t obase = (size_t)i * KNB * H;

#pragma unroll
    for (int t = 0; t < 3; ++t) {           // M-tiles: neighbor rows 16t..16t+15
        v8f c = {};
        const int row = 16 * t + m;
#pragma unroll
        for (int c4 = 0; c4 < 4; ++c4) {
            const int c0 = 32 * c4 + khalf;
            v16h Ah = pack16(*(const v8h*)(Ah_s + row * H + c0),
                             *(const v8h*)(Ah_s + row * H + c0 + 16));
            v16h Al = pack16(*(const v8h*)(Al_s + row * H + c0),
                             *(const v8h*)(Al_s + row * H + c0 + 16));
            c = wmma_acc(Ah, Bh[c4], c);
            c = wmma_acc(Ah, Bl[c4], c);
            c = wmma_acc(Al, Bh[c4], c);
        }
#pragma unroll
        for (int r = 0; r < 8; ++r) {
            const int kk = 16 * t + r + mofs;
            out[obase + (size_t)kk * H + o] = c[r] + bias;
        }
    }
}

extern "C" void kernel_launch(void* const* d_in, const int* in_sizes, int n_in,
                              void* d_out, int out_size, void* d_ws, size_t ws_size,
                              hipStream_t stream) {
    const float* hV   = (const float*)d_in[0];
    const int*   Eidx = (const int*)  d_in[1];
    const float* W1   = (const float*)d_in[2];
    const float* b1   = (const float*)d_in[3];
    const float* W2   = (const float*)d_in[4];
    const float* b2   = (const float*)d_in[5];
    const float* lnS  = (const float*)d_in[6];
    const float* lnB  = (const float*)d_in[7];
    float* out = (float*)d_out;

    // Workspace carve-up (all 16-byte aligned): 512 KB + 4*32 KB = 640 KB
    char* w = (char*)d_ws;
    float*    hv  = (float*)w;                                  // [1024][128] f32
    _Float16* W1h = (_Float16*)(w + (size_t)L * H * sizeof(float));
    _Float16* W1l = W1h + H * H;
    _Float16* W2h = W1l + H * H;
    _Float16* W2l = W2h + H * H;

    split_w_kernel<<<64, 256, 0, stream>>>(W1, W2, W1h, W1l, W2h, W2l);
    ln_gemm1_kernel<<<L / 16, 256, 0, stream>>>(hV, lnS, lnB, W1h, W1l, b1, hv);
    gemm2_kernel<<<L, 256, 0, stream>>>(hv, Eidx, W2h, W2l, b2, out);
}
